// ProjectedAdaptiveLogSoftmax_44573170597955
// MI455X (gfx1250) — compile-verified, matrix operand order checked by reading, not run on hardware
//
#include <hip/hip_runtime.h>
#include <hip/hip_bf16.h>

// ---------------------------------------------------------------------------
// Projected adaptive log-softmax, MI455X (gfx1250, wave32, WMMA bf16)
// ---------------------------------------------------------------------------
//   N=1024 rows, D_PROJ=1024
//   cluster 0 (head): vocab 19997 + 3 cluster logits -> padded 20000, d=1024
//   cluster 1: vocab 20000, d=256
//   cluster 2: vocab 160000, d=64
//   cluster 3: vocab 67738 -> padded 67744, d=16 -> padded K=32
// ---------------------------------------------------------------------------

typedef __attribute__((ext_vector_type(16))) __bf16 v16bf;
typedef __attribute__((ext_vector_type(8)))  float  v8f;

#define LDH 1376   // h row width: 1024 + 256 + 64 + 32 (tail3 K padded to 32)

// workspace byte offsets
#define O_HIDBF 0L                         // hidden bf16           1024*1024*2
#define O_PT    2097152L                   // pT bf16 (d x 1024)    1392640*2
#define O_HF32  4882432L                   // h fp32 [1024 x 1376]  1409024*4
#define O_HBF   10518528L                  // h bf16 [1024 x 1376]  1409024*2
#define O_WBF   13336576L                  // padded weights bf16   38007808*2
#define O_BIAS  89352192L                  // padded bias fp32      267744*4
#define O_PART  90423168L                  // partials (m,s)        262144*4

__device__ __forceinline__ unsigned short f2bf(float x) {
  unsigned u = __float_as_uint(x);
  unsigned r = u + 0x7fffu + ((u >> 16) & 1u);   // round-to-nearest-even
  return (unsigned short)(r >> 16);
}

// Load a 16-element bf16 WMMA fragment for one lane.
// p points at (row*ld + k0 + half*8); elements 0..7 there, 8..15 at +16.
__device__ __forceinline__ v16bf load_frag(const unsigned short* p) {
  union { uint4 u[2]; v16bf v; } f;
  f.u[0] = *(const uint4*)(p);
  f.u[1] = *(const uint4*)(p + 16);
  return f.v;
}

__device__ __forceinline__ v8f wmma_bf16(v16bf a, v16bf b, v8f c) {
  return __builtin_amdgcn_wmma_f32_16x16x32_bf16(false, a, false, b,
                                                 (short)0, c, false, false);
}

// --------------------------- conversion kernels -----------------------------

__global__ void k_cvt_f32_bf16(const float* __restrict__ src,
                               unsigned short* __restrict__ dst, long n) {
  long i = (long)blockIdx.x * blockDim.x + threadIdx.x;
  if (i < n) dst[i] = f2bf(src[i]);
}

// p is [1024, d] row-major; store transposed pT [d, 1024]
__global__ void k_cvt_pT(const float* __restrict__ src,
                         unsigned short* __restrict__ dst, int d) {
  long i = (long)blockIdx.x * blockDim.x + threadIdx.x;
  if (i >= (long)d * 1024) return;
  int n = (int)(i >> 10);
  int k = (int)(i & 1023);
  dst[i] = f2bf(src[(long)k * d + n]);
}

// padded weight conversion; src2 (optional) supplies rows [srcV, srcV+3)
__global__ void k_cvt_W(const float* __restrict__ src,
                        const float* __restrict__ src2,
                        unsigned short* __restrict__ dst,
                        int srcV, int srcD, long n, int padD) {
  long i = (long)blockIdx.x * blockDim.x + threadIdx.x;
  if (i >= n) return;
  long v = i / padD;
  int  k = (int)(i - v * padD);
  float x = 0.f;
  if (k < srcD) {
    if (v < srcV)             x = src[v * srcD + k];
    else if (src2 != nullptr) x = src2[(v - srcV) * (long)srcD + k];
  }
  dst[i] = f2bf(x);
}

__global__ void k_bias(const float* __restrict__ b, const float* __restrict__ b2,
                       float* __restrict__ dst, int srcV, int padV) {
  int v = blockIdx.x * blockDim.x + threadIdx.x;
  if (v >= padV) return;
  float x = -1e30f;                     // padding -> never contributes to lse
  if (v < srcV)                 x = b[v];
  else if (b2 && v < srcV + 3)  x = b2[v - srcV];
  dst[v] = x;
}

// zero bf16 K-pad (cols 1360..1375) of h for cluster 3
__global__ void k_zero_hpad(unsigned short* __restrict__ hb) {
  int i = blockIdx.x * blockDim.x + threadIdx.x;
  if (i >= 1024 * 16) return;
  int r = i >> 4, j = i & 15;
  hb[(long)r * LDH + 1360 + j] = 0;
}

// ------------------------------ projections ---------------------------------
// h_c = hidden_bf16 @ p_c_bf16  (K=1024, fp32 accum), 1 wave per 16x16 tile.
// grid = (64 row tiles, 85 col tiles), block = 32 (no partial waves -> EXEC=~0).

__global__ __launch_bounds__(32) void k_proj(char* __restrict__ ws) {
  const int PTOFF[4] = {0, 1048576, 1310720, 1376256};
  const int COFF[4]  = {0, 1024, 1280, 1344};
  const int lane = threadIdx.x & 31;
  const int half = lane >> 4;
  const int l16  = lane & 15;

  int ct = blockIdx.y;
  int cidx, n0;
  if      (ct < 64) { cidx = 0; n0 = ct * 16; }
  else if (ct < 80) { cidx = 1; n0 = (ct - 64) * 16; }
  else if (ct < 84) { cidx = 2; n0 = (ct - 80) * 16; }
  else              { cidx = 3; n0 = 0; }

  const unsigned short* hidbf = (const unsigned short*)(ws + O_HIDBF);
  const unsigned short* pT    = (const unsigned short*)(ws + O_PT) + PTOFF[cidx];

  const int r0 = blockIdx.x * 16;
  const unsigned short* arow = hidbf + (long)(r0 + l16) * 1024 + half * 8;
  const unsigned short* brow = pT    + (long)(n0 + l16) * 1024 + half * 8;

  v8f c = {};
#pragma unroll
  for (int kc = 0; kc < 32; ++kc) {
    v16bf a = load_frag(arow + kc * 32);
    v16bf b = load_frag(brow + kc * 32);
    c = wmma_bf16(a, b, c);
  }

  float*          hf = (float*)(ws + O_HF32);
  unsigned short* hb = (unsigned short*)(ws + O_HBF);
  const int col = COFF[cidx] + n0 + l16;
#pragma unroll
  for (int j = 0; j < 8; ++j) {
    int r = r0 + half * 8 + j;
    float v = c[j];
    hf[(long)r * LDH + col] = v;
    hb[(long)r * LDH + col] = f2bf(v);
  }
}

// ----------------------- main GEMM + streaming LSE --------------------------
// One instantiation per cluster (KSTEPS, LD compile-time -> fully unrolled,
// software-pipelined K loop). Each wave owns 32 rows x 32 vocab per tile step
// (2x2 register blocking: 2 A frags + 2 B frags -> 4 WMMAs per K chunk).
// grid = (32 vocab stripes, 4 row blocks), block = 256 (8 waves).
// Per-row online logsumexp; one (m,s) partial per row per stripe.

// combine one 16x16 logit tile (this lane: col l16, rows 8*half+j) into the
// running per-row (m, s) registers; cross-lane reduce over 16 lanes per half.
__device__ __forceinline__ void lse_update(const v8f& cacc, float bv,
                                           float* mrun, float* srun) {
#pragma unroll
  for (int j = 0; j < 8; ++j) {
    float t = cacc[j] + bv;
    float v = t;
    for (int m = 8; m >= 1; m >>= 1) v = fmaxf(v, __shfl_xor(v, m, 32));
    float e = __expf(t - v);
    for (int m = 8; m >= 1; m >>= 1) e += __shfl_xor(e, m, 32);
    float mn = fmaxf(mrun[j], v);
    srun[j] = srun[j] * __expf(mrun[j] - mn) + e * __expf(v - mn);
    mrun[j] = mn;
  }
}

template <int KSTEPS, int LD>
__global__ __launch_bounds__(256) void k_gemm_lse_t(
    const unsigned short* __restrict__ Wc, const float* __restrict__ bias,
    const unsigned short* __restrict__ hb, float* __restrict__ pm,
    float* __restrict__ ps, int nvt2) {
  const int lane = threadIdx.x & 31;
  const int wave = threadIdx.x >> 5;
  const int half = lane >> 4;
  const int l16  = lane & 15;
  const int r0   = (blockIdx.y * 8 + wave) * 32;

  float mrun[2][8], srun[2][8];
#pragma unroll
  for (int rt = 0; rt < 2; ++rt)
#pragma unroll
    for (int j = 0; j < 8; ++j) { mrun[rt][j] = -3.0e38f; srun[rt][j] = 0.f; }

  const unsigned short* arow0 = hb + (long)(r0 + l16) * LDH + half * 8;
  const unsigned short* arow1 = arow0 + 16L * LDH;

  for (int vt2 = blockIdx.x; vt2 < nvt2; vt2 += 32) {
    const int v0 = vt2 * 32;
    const unsigned short* brow0 = Wc + (long)(v0 + l16) * LD + half * 8;
    const unsigned short* brow1 = brow0 + 16L * LD;

    v8f a00 = {}, a01 = {}, a10 = {}, a11 = {};
#pragma unroll
    for (int kc = 0; kc < KSTEPS; ++kc) {
      v16bf fa0 = load_frag(arow0 + kc * 32);
      v16bf fa1 = load_frag(arow1 + kc * 32);
      v16bf fb0 = load_frag(brow0 + kc * 32);
      v16bf fb1 = load_frag(brow1 + kc * 32);
      a00 = wmma_bf16(fa0, fb0, a00);
      a01 = wmma_bf16(fa0, fb1, a01);
      a10 = wmma_bf16(fa1, fb0, a10);
      a11 = wmma_bf16(fa1, fb1, a11);
    }
    const float bv0 = bias[v0 + l16];
    const float bv1 = bias[v0 + 16 + l16];
    lse_update(a00, bv0, mrun[0], srun[0]);
    lse_update(a01, bv1, mrun[0], srun[0]);
    lse_update(a10, bv0, mrun[1], srun[1]);
    lse_update(a11, bv1, mrun[1], srun[1]);
  }

  if (l16 == 0) {
#pragma unroll
    for (int rt = 0; rt < 2; ++rt)
#pragma unroll
      for (int j = 0; j < 8; ++j) {
        int r = r0 + rt * 16 + half * 8 + j;
        pm[(long)blockIdx.x * 1024 + r] = mrun[rt][j];
        ps[(long)blockIdx.x * 1024 + r] = srun[rt][j];
      }
  }
}

// -------------------------------- finalize ----------------------------------
// one wave per row: combine 32 partials per cluster, gather target logit, NLL.

__global__ __launch_bounds__(256) void k_finalize(
    const char* __restrict__ ws, const int* __restrict__ tgt,
    const float* __restrict__ W0, const float* __restrict__ b0,
    const float* __restrict__ W1, const float* __restrict__ b1,
    const float* __restrict__ W2, const float* __restrict__ b2,
    const float* __restrict__ W3, const float* __restrict__ b3,
    const float* __restrict__ cw, const float* __restrict__ cb,
    float* __restrict__ out) {
  const int lane = threadIdx.x & 31;
  const int wave = threadIdx.x >> 5;
  const int r = blockIdx.x * 8 + wave;

  const float* part = (const float*)(ws + O_PART);
  const float* hf   = (const float*)(ws + O_HF32) + (long)r * LDH;

  float lse[4];
#pragma unroll
  for (int c = 0; c < 4; ++c) {
    float m = part[(long)(c * 2 + 0) * 32768 + (long)lane * 1024 + r];
    float s = part[(long)(c * 2 + 1) * 32768 + (long)lane * 1024 + r];
    float M = m;
    for (int x = 16; x >= 1; x >>= 1) M = fmaxf(M, __shfl_xor(M, x, 32));
    float e = s * __expf(m - M);
    for (int x = 16; x >= 1; x >>= 1) e += __shfl_xor(e, x, 32);
    lse[c] = M + __logf(e);
  }

  const int t = tgt[r];
  float lp;
  if (t < 19997) {
    float acc = 0.f;
    for (int k = lane; k < 1024; k += 32) acc += hf[k] * W0[(long)t * 1024 + k];
    for (int x = 16; x >= 1; x >>= 1) acc += __shfl_xor(acc, x, 32);
    lp = acc + b0[t] - lse[0];
  } else {
    int c, lo;
    if      (t < 39997)  { c = 1; lo = 19997; }
    else if (t < 199997) { c = 2; lo = 39997; }
    else                 { c = 3; lo = 199997; }
    const int rel = t - lo;
    const float* Wt = (c == 1) ? W1 : (c == 2) ? W2 : W3;
    const float* bt = (c == 1) ? b1 : (c == 2) ? b2 : b3;
    const int dc = (c == 1) ? 256 : (c == 2) ? 64 : 16;
    const int co = (c == 1) ? 1024 : (c == 2) ? 1280 : 1344;
    // head cluster logit
    float accc = 0.f;
    for (int k = lane; k < 1024; k += 32) accc += hf[k] * cw[(long)(c - 1) * 1024 + k];
    for (int x = 16; x >= 1; x >>= 1) accc += __shfl_xor(accc, x, 32);
    // tail target logit
    float acct = 0.f;
    for (int k = lane; k < dc; k += 32) acct += hf[co + k] * Wt[(long)rel * dc + k];
    for (int x = 16; x >= 1; x >>= 1) acct += __shfl_xor(acct, x, 32);
    lp = (accc + cb[c - 1] - lse[0]) + (acct + bt[rel] - lse[c]);
  }
  if (lane == 0) out[r] = -lp;
}

// -------------------------------- launcher ----------------------------------

extern "C" void kernel_launch(void* const* d_in, const int* in_sizes, int n_in,
                              void* d_out, int out_size, void* d_ws, size_t ws_size,
                              hipStream_t stream) {
  const float* hidden = (const float*)d_in[0];
  const int*   target = (const int*)d_in[1];
  const float* W0 = (const float*)d_in[2];
  const float* b0 = (const float*)d_in[3];
  const float* p0 = (const float*)d_in[4];
  const float* W1 = (const float*)d_in[5];
  const float* b1 = (const float*)d_in[6];
  const float* p1 = (const float*)d_in[7];
  const float* W2 = (const float*)d_in[8];
  const float* b2 = (const float*)d_in[9];
  const float* p2 = (const float*)d_in[10];
  const float* W3 = (const float*)d_in[11];
  const float* b3 = (const float*)d_in[12];
  const float* p3 = (const float*)d_in[13];
  const float* cw = (const float*)d_in[14];
  const float* cb = (const float*)d_in[15];
  float* out = (float*)d_out;
  char*  ws  = (char*)d_ws;

  const int TB = 256;
  auto nb = [](long n, int b) { return (unsigned)((n + b - 1) / b); };

  // 1) hidden -> bf16
  k_cvt_f32_bf16<<<nb(1024L * 1024, TB), TB, 0, stream>>>(
      hidden, (unsigned short*)(ws + O_HIDBF), 1024L * 1024);

  // 2) p_i -> transposed bf16
  const long PTOFF[4] = {0, 1048576, 1310720, 1376256};
  const float* ps_[4] = {p0, p1, p2, p3};
  const int    pd_[4] = {1024, 256, 64, 16};
  for (int c = 0; c < 4; ++c)
    k_cvt_pT<<<nb((long)pd_[c] * 1024, TB), TB, 0, stream>>>(
        ps_[c], (unsigned short*)(ws + O_PT) + PTOFF[c], pd_[c]);

  // 3) padded weights -> bf16
  const long  WOFF[4]  = {0L, 20480000L, 25600000L, 35840000L};
  const float* wsrc[4] = {W0, W1, W2, W3};
  const float* w2_[4]  = {cw, nullptr, nullptr, nullptr};
  const int   srcV[4]  = {19997, 20000, 160000, 67738};
  const int   srcD[4]  = {1024, 256, 64, 16};
  const int   padV[4]  = {20000, 20000, 160000, 67744};
  const int   padD[4]  = {1024, 256, 64, 32};
  for (int c = 0; c < 4; ++c) {
    long n = (long)padV[c] * padD[c];
    k_cvt_W<<<nb(n, TB), TB, 0, stream>>>(
        wsrc[c], w2_[c], (unsigned short*)(ws + O_WBF) + WOFF[c],
        srcV[c], srcD[c], n, padD[c]);
  }

  // 4) padded biases
  const long BOFF[4]   = {0, 20000, 40000, 200000};
  const float* bsrc[4] = {b0, b1, b2, b3};
  const float* b2_[4]  = {cb, nullptr, nullptr, nullptr};
  for (int c = 0; c < 4; ++c)
    k_bias<<<nb(padV[c], TB), TB, 0, stream>>>(
        bsrc[c], b2_[c], (float*)(ws + O_BIAS) + BOFF[c], srcV[c], padV[c]);

  // 5) zero bf16 K-pad of h for cluster 3
  k_zero_hpad<<<nb(1024L * 16, TB), TB, 0, stream>>>((unsigned short*)(ws + O_HBF));

  // 6) projections (WMMA): 64 row tiles x 85 col tiles, 1 wave each
  k_proj<<<dim3(64, 85, 1), 32, 0, stream>>>(ws);

  // 7) main GEMM + streaming logsumexp (WMMA), specialized per cluster
  {
    const unsigned short* Wb  = (const unsigned short*)(ws + O_WBF);
    const float*          bb  = (const float*)(ws + O_BIAS);
    const unsigned short* hbb = (const unsigned short*)(ws + O_HBF);
    float*                pb  = (float*)(ws + O_PART);
    const dim3 g(32, 4, 1);
    // cluster 0: head, d=1024
    k_gemm_lse_t<32, 1024><<<g, 256, 0, stream>>>(
        Wb + 0L,        bb + 0,      hbb + 0,    pb + 0L * 32768, pb + 1L * 32768, 625);
    // cluster 1: d=256
    k_gemm_lse_t<8, 256><<<g, 256, 0, stream>>>(
        Wb + 20480000L, bb + 20000,  hbb + 1024, pb + 2L * 32768, pb + 3L * 32768, 625);
    // cluster 2: d=64
    k_gemm_lse_t<2, 64><<<g, 256, 0, stream>>>(
        Wb + 25600000L, bb + 40000,  hbb + 1280, pb + 4L * 32768, pb + 5L * 32768, 5000);
    // cluster 3: d=16 (padded K=32)
    k_gemm_lse_t<1, 32><<<g, 256, 0, stream>>>(
        Wb + 35840000L, bb + 200000, hbb + 1344, pb + 6L * 32768, pb + 7L * 32768, 2117);
  }

  // 8) combine partials + gather target logits -> NLL
  k_finalize<<<128, 256, 0, stream>>>(ws, target, W0, b0, W1, b1, W2, b2,
                                      W3, b3, cw, cb, out);
}
